// SelfAttention3D_30193620090937
// MI455X (gfx1250) — matrix operand order repaired and checked by reading, not compiled, not run on hardware
//
#include <hip/hip_runtime.h>
#include <hip/hip_bf16.h>

typedef __attribute__((ext_vector_type(16))) _Float16 v16h;
typedef __attribute__((ext_vector_type(8)))  float    v8f;
typedef unsigned int u32x4 __attribute__((ext_vector_type(4)));
typedef int          i32x8 __attribute__((ext_vector_type(8)));
typedef int          i32x4 __attribute__((ext_vector_type(4)));

#define CH  256   // channels
#define CQn 32    // q/k channels
#define NN  4096  // spatial positions
#define BB  4     // batch
#define LOG2E 1.44269504088896340736f

// dynamic LDS carve-up (bytes)
#define PSTRIDE   72                          // halfs per row: 144B, 16B aligned, 36-dword bank stride
#define PLDS_SZ   (8 * 16 * PSTRIDE * 2)      // 18432: per-wave P transpose tiles
#define VTILE_SZ  (CH * PSTRIDE * 2)          // 36864: one 256x64 f16 V chunk (padded rows)
#define SMEM_SZ   (PLDS_SZ + 2 * VTILE_SZ)    // 92160

// ---------------------------------------------------------------------------
// Projection kernel: q = (Wq x + bq) * log2e  (f16 [B,N,32]) so the attention
// kernel can use native exp2; k transposed to [B,N,32]; v kept [B,C,N].
// ---------------------------------------------------------------------------
__global__ __launch_bounds__(256) void proj_kernel(
    const float* __restrict__ x,
    const float* __restrict__ wq, const float* __restrict__ bq,
    const float* __restrict__ wk, const float* __restrict__ bk,
    const float* __restrict__ wv, const float* __restrict__ bv,
    _Float16* __restrict__ qh, _Float16* __restrict__ kh,
    _Float16* __restrict__ vh)
{
    const int n   = blockIdx.x * blockDim.x + threadIdx.x;
    const int row = blockIdx.y;
    const int b   = blockIdx.z;
    const float* xb = x + ((size_t)b * CH) * NN + n;

    const float* w; float bias; int o;
    if (row < CQn)          { o = row;           w = wq + (size_t)o * CH; bias = bq[o]; }
    else if (row < 2 * CQn) { o = row - CQn;     w = wk + (size_t)o * CH; bias = bk[o]; }
    else                    { o = row - 2 * CQn; w = wv + (size_t)o * CH; bias = bv[o]; }

    float acc = bias;
    #pragma unroll 8
    for (int c = 0; c < CH; ++c)
        acc = fmaf(w[c], xb[(size_t)c * NN], acc);

    if (row < CQn)          qh[((size_t)b * NN + n) * CQn + o] = (_Float16)(acc * LOG2E);
    else if (row < 2 * CQn) kh[((size_t)b * NN + n) * CQn + o] = (_Float16)acc;
    else                    vh[((size_t)b * CH + o) * NN + n]  = (_Float16)acc;
}

// ---------------------------------------------------------------------------
// WMMA operand loaders (CDNA5 16-bit layouts, wave32).
// ---------------------------------------------------------------------------
__device__ __forceinline__ v16h load_a_f16(const _Float16* base, size_t strideHalf, int lane) {
    const int r = lane & 15, h = lane >> 4;
    const _Float16* row = base + (size_t)r * strideHalf;
    union { uint4 u[2]; v16h v; } U;
    U.u[0] = *(const uint4*)(row + h * 8);
    U.u[1] = *(const uint4*)(row + 16 + h * 8);
    return U.v;
}

__device__ __forceinline__ v16h load_b_f16(const _Float16* base, size_t strideHalf, int lane) {
    const int c = lane & 15, h = lane >> 4;
    const _Float16* row = base + (size_t)c * strideHalf;
    union { uint4 u[2]; v16h v; } U;
    U.u[0] = *(const uint4*)(row + h * 8);
    U.u[1] = *(const uint4*)(row + 16 + h * 8);
    return U.v;
}

// DPP16 max-reduce across each 16-lane half (pure VALU).
#define DPP_MAXSTEP(v, ctrl)                                                     \
    (v) = fmaxf((v), __builtin_bit_cast(float, __builtin_amdgcn_update_dpp(      \
              0, __builtin_bit_cast(int, (v)), (ctrl), 0xF, 0xF, true)))

__device__ __forceinline__ float rowmax16_dpp(float v) {
    DPP_MAXSTEP(v, 0xB1);   // quad_perm [1,0,3,2]
    DPP_MAXSTEP(v, 0x4E);   // quad_perm [2,3,0,1]
    DPP_MAXSTEP(v, 0x141);  // row_half_mirror
    DPP_MAXSTEP(v, 0x140);  // row_mirror
    return v;
}

// ---------------------------------------------------------------------------
// Tensor Data Mover: DMA one V chunk (2D tile: 64 f16 along m x 256 channel
// rows, row stride N) from global into LDS with 16B padding per 128B row
// (-> 144B LDS row stride == PSTRIDE halfs). D# packed per CDNA5 ISA ch.8.
// ---------------------------------------------------------------------------
__device__ __forceinline__ void tdm_load_vchunk(const _Float16* gsrc, unsigned ldsOff) {
    const unsigned long long ga = (unsigned long long)(uintptr_t)gsrc;
    u32x4 g0;
    g0[0] = 1u;                                   // count=1 valid descriptor
    g0[1] = ldsOff;                               // lds_addr (bytes)
    g0[2] = (unsigned)ga;                         // global_addr[31:0]
    g0[3] = (unsigned)(ga >> 32) | 0x80000000u;   // global_addr[56:32] | type=2<<30
    i32x8 g1;
    g1[0] = (1 << 16)        // data_size = 2 bytes
          | (1 << 20)        // pad_enable
          | (4 << 22)        // pad_interval: every 32 dwords (one 128B row)
          | (3 << 25);       // pad_amount: 4 dwords (16B) -> 144B row stride
    g1[1] = (int)((NN & 0xFFFF) << 16);                              // tensor_dim0 lo16
    g1[2] = (int)(((NN >> 16) & 0xFFFF) | ((CH & 0xFFFF) << 16));    // dim0 hi16 | dim1 lo16
    g1[3] = (int)(64 << 16);                                         // dim1 hi16=0 | tile_dim0=64
    g1[4] = CH;                                                      // tile_dim1=256, tile_dim2=0
    g1[5] = NN;                                                      // tensor_dim0_stride lo32
    g1[6] = 0;                                                       // stride hi | dim1_stride lo
    g1[7] = 0;
    const i32x4 gz = (i32x4){};
#if __clang_major__ >= 23
    const i32x8 gz8 = (i32x8){};
    __builtin_amdgcn_tensor_load_to_lds(g0, g1, gz, gz, gz8, 0);
#else
    __builtin_amdgcn_tensor_load_to_lds(g0, g1, gz, gz, 0);
#endif
}

// ---------------------------------------------------------------------------
// Flash-attention kernel. 8 waves/block, 16 query rows/wave. Per 64-key chunk:
//   S(16x64) = Q @ K^T   -> 4x v_wmma_f32_16x16x32_f16
//   rowsum  += P @ ones  -> 2x v_wmma   (matrix-unit softmax row sums)
//   O       += P @ V^T   -> 32x v_wmma  (V read from the TDM-staged LDS tile)
// V chunks are DMA'd by the Tensor Data Mover once per WORKGROUP (not per
// wave), double-buffered and overlapped with compute; wave 0 owns TENSORcnt
// and publishes completion via the workgroup barrier.
// ---------------------------------------------------------------------------
__global__ __launch_bounds__(256) void attn_kernel(
    const _Float16* __restrict__ qh, const _Float16* __restrict__ kh,
    const _Float16* __restrict__ vh, const float* __restrict__ x,
    const float* __restrict__ gamma, float* __restrict__ out)
{
    extern __shared__ char smem[];
    _Float16* plds  = (_Float16*)smem;
    _Float16* vbuf0 = (_Float16*)(smem + PLDS_SZ);
    _Float16* vbuf1 = (_Float16*)(smem + PLDS_SZ + VTILE_SZ);
    // LDS byte offsets for the TDM descriptor: dynamic LDS starts right after
    // static LDS (we have none) -> groupstaticsize + carve offset. No
    // pointer->int addrspacecast needed.
    const unsigned dynBase = (unsigned)__builtin_amdgcn_groupstaticsize();
    const unsigned vOff0 = dynBase + PLDS_SZ;
    const unsigned vOff1 = dynBase + PLDS_SZ + VTILE_SZ;

    const int tid  = threadIdx.x;
    const int wave = tid >> 5, lane = tid & 31;
    const int col  = lane & 15, half = lane >> 4;
    const int b    = blockIdx.y;
    const int qbase = (blockIdx.x * 8 + wave) * 16;

    const _Float16* qb = qh + ((size_t)b * NN + qbase) * CQn;
    const _Float16* kb = kh + (size_t)b * NN * CQn;
    const _Float16* vb = vh + (size_t)b * CH * NN;
    _Float16* pw = plds + wave * (16 * PSTRIDE);

    const v16h qA = load_a_f16(qb, CQn, lane);

    v16h ones;
    #pragma unroll
    for (int i = 0; i < 16; ++i) ones[i] = (_Float16)1.0f;

    v8f acc[16];
    #pragma unroll
    for (int t = 0; t < 16; ++t) acc[t] = (v8f){};
    v8f accsum = (v8f){};
    float rmax = -1.0e30f;  // wave-uniform running max (log2 domain)

    // prime the pipeline: DMA chunk 0 into buffer 0
    if (wave == 0) tdm_load_vchunk(vb, vOff0);

    #pragma unroll 1
    for (int m0 = 0; m0 < NN; m0 += 64) {
        const int cur = (m0 >> 6) & 1;

        // issue next chunk's DMA (into the buffer freed by the trailing
        // barrier of the previous iteration), then wait for current chunk
        if (wave == 0) {
            if (m0 + 64 < NN) {
                tdm_load_vchunk(vb + (m0 + 64), cur ? vOff0 : vOff1);
                __builtin_amdgcn_s_wait_tensorcnt(1);
            } else {
                __builtin_amdgcn_s_wait_tensorcnt(0);
            }
        }
        __syncthreads();  // publish vbuf[cur] to all waves
        const _Float16* vt = cur ? vbuf1 : vbuf0;

        // ---- S = Q @ K^T for 64 keys (log2-domain logits) ----
        v8f s[4];
        #pragma unroll
        for (int j = 0; j < 4; ++j) {
            const v16h kB = load_b_f16(kb + (size_t)(m0 + 16 * j) * CQn, CQn, lane);
            v8f z = (v8f){};
            s[j] = __builtin_amdgcn_wmma_f32_16x16x32_f16(false, qA, false, kB, (short)0, z, false, false);
        }

        // ---- wave-uniform chunk max: per-lane tree, DPP16, cross-half ----
        float cm = s[0][0];
        #pragma unroll
        for (int j = 0; j < 4; ++j)
            #pragma unroll
            for (int r = 0; r < 8; ++r)
                if (j || r) cm = fmaxf(cm, s[j][r]);
        cm = rowmax16_dpp(cm);
        cm = fmaxf(cm, __shfl_xor(cm, 16, 32));
        const float tmax = __builtin_bit_cast(float,
            __builtin_amdgcn_readfirstlane(__builtin_bit_cast(int, cm)));

        // ---- uniform-branch rescale, only when the max grows ----
        if (tmax > rmax) {
            const float scf = __builtin_exp2f(rmax - tmax);
            rmax = tmax;
            #pragma unroll
            for (int r = 0; r < 8; ++r) accsum[r] *= scf;
            #pragma unroll
            for (int t = 0; t < 16; ++t)
                #pragma unroll
                for (int r = 0; r < 8; ++r)
                    acc[t][r] *= scf;
        }

        // ---- P = exp2(S - rmax), staged to per-wave LDS tile ----
        #pragma unroll
        for (int r = 0; r < 8; ++r) {
            const int prow = (r + 8 * half) * PSTRIDE + col;
            #pragma unroll
            for (int j = 0; j < 4; ++j)
                pw[prow + 16 * j] = (_Float16)__builtin_exp2f(s[j][r] - rmax);
        }

        // ---- reload P in A layout; rowsum += P @ 1, O += P @ V^T ----
        const v16h pA0 = load_a_f16(pw,      PSTRIDE, lane);
        const v16h pA1 = load_a_f16(pw + 32, PSTRIDE, lane);
        accsum = __builtin_amdgcn_wmma_f32_16x16x32_f16(false, pA0, false, ones, (short)0, accsum, false, false);
        accsum = __builtin_amdgcn_wmma_f32_16x16x32_f16(false, pA1, false, ones, (short)0, accsum, false, false);
        #pragma unroll
        for (int t = 0; t < 16; ++t) {
            const v16h vB0 = load_b_f16(vt + (size_t)(t * 16) * PSTRIDE,      PSTRIDE, lane);
            const v16h vB1 = load_b_f16(vt + (size_t)(t * 16) * PSTRIDE + 32, PSTRIDE, lane);
            acc[t] = __builtin_amdgcn_wmma_f32_16x16x32_f16(false, pA0, false, vB0, (short)0, acc[t], false, false);
            acc[t] = __builtin_amdgcn_wmma_f32_16x16x32_f16(false, pA1, false, vB1, (short)0, acc[t], false, false);
        }
        __syncthreads();  // all waves done with vbuf[cur] before it is re-DMA'd
    }

    // ---- epilogue: out = gamma * (O / rowsum) + x ----
    const float g = gamma[0];
    #pragma unroll
    for (int r = 0; r < 8; ++r) {
        const float inv = 1.0f / accsum[r];
        const int n = qbase + r + 8 * half;
        #pragma unroll
        for (int t = 0; t < 16; ++t) {
            const int c = t * 16 + col;
            const size_t idx = ((size_t)b * CH + c) * NN + n;
            out[idx] = g * acc[t][r] * inv + x[idx];
        }
    }
}

// ---------------------------------------------------------------------------
extern "C" void kernel_launch(void* const* d_in, const int* in_sizes, int n_in,
                              void* d_out, int out_size, void* d_ws, size_t ws_size,
                              hipStream_t stream) {
    (void)in_sizes; (void)n_in; (void)out_size; (void)ws_size;
    const float* x     = (const float*)d_in[0];
    const float* wq    = (const float*)d_in[1];
    const float* bq    = (const float*)d_in[2];
    const float* wk    = (const float*)d_in[3];
    const float* bk    = (const float*)d_in[4];
    const float* wv    = (const float*)d_in[5];
    const float* bv    = (const float*)d_in[6];
    const float* gamma = (const float*)d_in[7];
    float* out = (float*)d_out;

    // workspace layout (f16): Q [B,N,32] | K^T [B,N,32] | V [B,C,N]  (~10 MB)
    _Float16* qh = (_Float16*)d_ws;
    _Float16* kh = qh + (size_t)BB * NN * CQn;
    _Float16* vh = kh + (size_t)BB * NN * CQn;

    dim3 pgrid(NN / 256, 2 * CQn + CH, BB);
    proj_kernel<<<pgrid, 256, 0, stream>>>(x, wq, bq, wk, bk, wv, bv, qh, kh, vh);

    // 90 KB dynamic LDS (WGP has 320 KB); raise the runtime cap.
    (void)hipFuncSetAttribute((const void*)attn_kernel,
                              hipFuncAttributeMaxDynamicSharedMemorySize, SMEM_SZ);
    dim3 agrid(NN / 128, BB);  // 8 waves/block * 16 query rows = 128 rows/block
    attn_kernel<<<agrid, 256, SMEM_SZ, stream>>>(qh, kh, vh, x, gamma, out);
}
